// PointNetPlusPlusEncoder_77713138254056
// MI455X (gfx1250) — matrix operand order repaired and checked by reading, not compiled, not run on hardware
//
#include <hip/hip_runtime.h>
#include <hip/hip_bf16.h>

typedef __attribute__((ext_vector_type(16))) _Float16 v16h;
typedef __attribute__((ext_vector_type(8)))  float    v8f;
typedef __attribute__((ext_vector_type(8)))  unsigned int v8u;
typedef unsigned long long u64;

#define PI_F 3.14159265358979323846f

// ---------------- wave32 cross-lane reductions (no barriers) ----------------
__device__ inline u64 wave_max_u64(u64 v) {
  #pragma unroll
  for (int s = 16; s > 0; s >>= 1) {
    u64 o = __shfl_xor(v, s, 32);
    if (o > v) v = o;
  }
  return v;
}
__device__ inline u64 wave_min_u64(u64 v) {
  #pragma unroll
  for (int s = 16; s > 0; s >>= 1) {
    u64 o = __shfl_xor(v, s, 32);
    if (o < v) v = o;
  }
  return v;
}

// ---------------------------------------------------------------------------
// Weight prep: W [K x N] f32 row-major  ->  Wt [N x K_pad] f16 (zero-padded K)
// ---------------------------------------------------------------------------
__global__ void prep_weight(const float* __restrict__ W, _Float16* __restrict__ Wt,
                            int K, int N, int Kp) {
  int i = blockIdx.x * blockDim.x + threadIdx.x;
  int total = N * Kp;
  if (i >= total) return;
  int n = i / Kp, kk = i % Kp;
  Wt[i] = (kk < K) ? (_Float16)W[(size_t)kk * N + n] : (_Float16)0.f;
}

// ---------------------------------------------------------------------------
// Farthest point sampling. One block per batch, dists in LDS, shfl reductions.
// argmax ties -> lowest index (matches jnp.argmax) via key = bits<<32 | ~i.
// ---------------------------------------------------------------------------
__global__ void fps_kernel(const float* __restrict__ pts, int N, int m,
                           int* __restrict__ outIdx) {
  extern __shared__ unsigned char smem[];
  float* d      = (float*)smem;
  u64*  partials = (u64*)(smem + (size_t)N * 4);
  const int b = blockIdx.x, t = threadIdx.x, bd = blockDim.x;
  const int lane = t & 31, w = t >> 5, nw = bd >> 5;
  const float* P = pts + (size_t)b * N * 3;
  float x0 = P[0], y0 = P[1], z0 = P[2];
  for (int i = t; i < N; i += bd) {
    float dx = P[i*3] - x0, dy = P[i*3+1] - y0, dz = P[i*3+2] - z0;
    d[i] = dx*dx + dy*dy + dz*dz;
  }
  if (t == 0) outIdx[(size_t)b * m] = 0;
  __syncthreads();
  for (int j = 1; j < m; ++j) {
    u64 best = 0;
    for (int i = t; i < N; i += bd) {
      unsigned int bits = __float_as_uint(d[i]);   // d >= 0: bit order == value order
      u64 key = ((u64)bits << 32) | (u64)(0xFFFFFFFFu - (unsigned)i);
      if (key > best) best = key;
    }
    best = wave_max_u64(best);
    if (lane == 0) partials[w] = best;
    __syncthreads();
    if (w == 0) {
      u64 v = (lane < nw) ? partials[lane] : 0;
      v = wave_max_u64(v);
      if (lane == 0) partials[0] = v;
    }
    __syncthreads();
    u64 sel = partials[0];
    int nxt = (int)(0xFFFFFFFFu - (unsigned)(sel & 0xFFFFFFFFull));
    if (t == 0) outIdx[(size_t)b * m + j] = nxt;
    float xn = P[nxt*3], yn = P[nxt*3+1], zn = P[nxt*3+2];
    for (int i = t; i < N; i += bd) {
      float dx = P[i*3] - xn, dy = P[i*3+1] - yn, dz = P[i*3+2] - zn;
      float dd = dx*dx + dy*dy + dz*dz;
      if (dd < d[i]) d[i] = dd;
    }
    __syncthreads();   // protects partials before next round's writes
  }
}

// ---------------------------------------------------------------------------
// Gather centers: out[b,i,:] = pts[b, idx[b,i], :]
// ---------------------------------------------------------------------------
__global__ void gather_centers(const float* __restrict__ pts, const int* __restrict__ idx,
                               float* __restrict__ out, int N, int m, int total) {
  int i = blockIdx.x * blockDim.x + threadIdx.x;
  if (i >= total) return;
  int c = i % 3, rest = i / 3;
  int ci = rest % m, b = rest / m;
  out[i] = pts[((size_t)b * N + idx[(size_t)b * m + ci]) * 3 + c];
}

// ---------------------------------------------------------------------------
// KNN: one block per center. Phase 1 computes order-mapped distance bits into
// LDS once; phase 2 runs k rounds of strictly-increasing lexicographic min
// over keys (ord<<32 | i) rebuilt from LDS. Matches top_k(-d) incl. tie order.
// ---------------------------------------------------------------------------
__global__ void knn_kernel(const float* __restrict__ pts, const float* __restrict__ centers,
                           int N, int m, int k, int* __restrict__ outIdx) {
  extern __shared__ unsigned char smem[];
  unsigned int* ords = (unsigned int*)smem;
  u64* partials      = (u64*)(smem + (size_t)N * 4);
  const int cid = blockIdx.x;             // global center id = b*m + ci
  const int b = cid / m;
  const float* P = pts + (size_t)b * N * 3;
  const float* C = centers + (size_t)cid * 3;
  float cx = C[0], cy = C[1], cz = C[2];
  float cc = cx*cx + cy*cy + cz*cz;
  const int t = threadIdx.x, bd = blockDim.x;
  const int lane = t & 31, w = t >> 5, nw = bd >> 5;
  for (int i = t; i < N; i += bd) {
    float px = P[i*3], py = P[i*3+1], pz = P[i*3+2];
    float dd = cc + px*px + py*py + pz*pz - 2.f*(cx*px + cy*py + cz*pz);
    unsigned int bits = __float_as_uint(dd);
    ords[i] = (bits & 0x80000000u) ? ~bits : (bits | 0x80000000u);
  }
  __syncthreads();
  u64 prev = 0;
  for (int j = 0; j < k; ++j) {
    u64 best = 0xFFFFFFFFFFFFFFFFull;
    for (int i = t; i < N; i += bd) {
      u64 key = ((u64)ords[i] << 32) | (u64)(unsigned)i;
      if (key > prev && key < best) best = key;
    }
    best = wave_min_u64(best);
    if (lane == 0) partials[w] = best;
    __syncthreads();
    if (w == 0) {
      u64 v = (lane < nw) ? partials[lane] : 0xFFFFFFFFFFFFFFFFull;
      v = wave_min_u64(v);
      if (lane == 0) partials[0] = v;
    }
    __syncthreads();
    u64 sel = partials[0];
    prev = sel;
    if (t == 0) outIdx[(size_t)cid * k + j] = (int)(sel & 0xFFFFFFFFull);
    __syncthreads();   // protects partials before next round's writes
  }
}

// ---------------------------------------------------------------------------
// SA1 edge features: 63 dims (diff + NeRF enc) -> f16 [rows x 64], col 63 = 0
// ---------------------------------------------------------------------------
__global__ void edge1_kernel(const float* __restrict__ pos, const float* __restrict__ centers1,
                             const int* __restrict__ nbr, _Float16* __restrict__ edge,
                             int total) {
  int row = blockIdx.x * blockDim.x + threadIdx.x;
  if (row >= total) return;
  int c = row >> 5;                 // center id = b*512 + ci
  int b = c >> 9;                   // /512
  int p = nbr[row];
  const float* P = pos + ((size_t)b * 8192 + p) * 3;
  const float* C = centers1 + (size_t)c * 3;
  float dv[3] = { P[0]-C[0], P[1]-C[1], P[2]-C[2] };
  _Float16* e = edge + (size_t)row * 64;
  e[0] = (_Float16)dv[0]; e[1] = (_Float16)dv[1]; e[2] = (_Float16)dv[2];
  #pragma unroll
  for (int ci = 0; ci < 3; ++ci) {
    #pragma unroll
    for (int l = 0; l < 10; ++l) {
      float s = dv[ci] * ((float)(1 << l) * PI_F);
      e[3 + ci*20 + l*2]     = (_Float16)__sinf(s);
      e[3 + ci*20 + l*2 + 1] = (_Float16)__cosf(s);
    }
  }
  e[63] = (_Float16)0.f;
}

// ---------------------------------------------------------------------------
// SA2 edge features: 319 dims (gathered x1[256] + diff/enc[63]) -> [rows x 320]
// One block (320 threads) per row.
// ---------------------------------------------------------------------------
__global__ void edge2_kernel(const _Float16* __restrict__ x1, const float* __restrict__ centers1,
                             const float* __restrict__ centers2, const int* __restrict__ nbr2,
                             _Float16* __restrict__ edge) {
  int row = blockIdx.x;              // (b*64 + ci)*32 + kk
  int t = threadIdx.x;               // 0..319
  int c2 = row >> 5;                 // global center2 id
  int b = c2 >> 6;                   // /64
  int p = nbr2[row];                 // index into this batch's 512 centers1
  _Float16* e = edge + (size_t)row * 320;
  if (t < 256) {
    e[t] = x1[((size_t)b * 512 + p) * 256 + t];
  } else if (t < 319) {
    const float* P = centers1 + ((size_t)b * 512 + p) * 3;
    const float* C = centers2 + (size_t)c2 * 3;
    float dv[3] = { P[0]-C[0], P[1]-C[1], P[2]-C[2] };
    int f = t - 256;                 // 0..62
    float val;
    if (f < 3) {
      val = dv[f];
    } else {
      int g = f - 3, ci = g / 20, rem = g % 20, l = rem >> 1;
      float s = dv[ci] * ((float)(1 << l) * PI_F);
      val = (rem & 1) ? __cosf(s) : __sinf(s);
    }
    e[t] = (_Float16)val;
  } else {
    e[t] = (_Float16)0.f;
  }
}

// ---------------------------------------------------------------------------
// WMMA GEMM: out[M x N] = A[M x Kp](f16) * Wt[N x Kp]^T(f16) + bias, opt ReLU.
// Each wave computes a 16x64 output strip (4 accumulators) so the streaming A
// fragment is loaded once per K-chunk and reused by 4 WMMAs. 4 waves/block.
// Requires N % 64 == 0 (true for all layers: 128/256/512).
// ---------------------------------------------------------------------------
__global__ __launch_bounds__(128) void gemm_wmma(
    const _Float16* __restrict__ A, const _Float16* __restrict__ Wt,
    const float* __restrict__ bias,
    _Float16* __restrict__ outH, float* __restrict__ outF,
    int M, int N, int Kp, int relu) {
  const int lane = threadIdx.x & 31;
  const int wave = threadIdx.x >> 5;
  const int groupsN = N >> 6;                    // strips of 64 columns
  const int groups  = (M >> 4) * groupsN;
  const int g = blockIdx.x * 4 + wave;
  if (g >= groups) return;
  const int tm = g / groupsN, tg = g % groupsN;
  const int r  = lane & 15;                      // row within tile (A), col (B/C)
  const int hs = lane >> 4;                      // half-select

  const _Float16* arow = A  + (size_t)(tm * 16 + r) * Kp;
  const _Float16* brow = Wt + (size_t)(tg * 64 + r) * Kp;
  const size_t bstep = (size_t)16 * Kp;

  v8f acc[4];
  #pragma unroll
  for (int j = 0; j < 4; ++j) acc[j] = (v8f){0.f,0.f,0.f,0.f,0.f,0.f,0.f,0.f};

  for (int k0 = 0; k0 < Kp; k0 += 32) {
    union { v8u u; v16h h; } ua;
    #pragma unroll
    for (int v = 0; v < 8; ++v) {
      // A: VGPR v<4 -> K=2v,2v+1 ; v>=4 -> K=16+2(v-4) ; lanes 16..31 add +8
      int ka = k0 + ((v & 3) << 1) + ((v & 4) << 2) + (hs << 3);
      ua.u[v] = *(const unsigned int*)(arow + ka);
    }
    #pragma unroll
    for (int j = 0; j < 4; ++j) {
      union { v8u u; v16h h; } ub;
      const _Float16* bp = brow + (size_t)j * bstep;
      #pragma unroll
      for (int v = 0; v < 8; ++v) {
        // B: VGPR v<4 -> K=2v,2v+1 ; v>=4 -> K=16+2(v-4) ; lanes 16..31 add +8
        int kb = k0 + ((v & 3) << 1) + ((v & 4) << 2) + (hs << 3);
        ub.u[v] = *(const unsigned int*)(bp + kb);
      }
      acc[j] = __builtin_amdgcn_wmma_f32_16x16x32_f16(
          false, ua.h, false, ub.h, (short)0, acc[j], false, false);
    }
  }
  #pragma unroll
  for (int j = 0; j < 4; ++j) {
    const int col = tg * 64 + j * 16 + r;
    float bv = bias ? bias[col] : 0.f;
    #pragma unroll
    for (int i = 0; i < 8; ++i) {
      int rowo = tm * 16 + i + hs * 8;
      float v = acc[j][i] + bv;
      if (relu) v = fmaxf(v, 0.f);
      if (outH) outH[(size_t)rowo * N + col] = (_Float16)v;
      else      outF[(size_t)rowo * N + col] = v;
    }
  }
}

// ---------------------------------------------------------------------------
// Max over the k=32 neighbor axis: in [centers*32 x C] -> out [centers x C]
// ---------------------------------------------------------------------------
__global__ void maxpool_kernel(const _Float16* __restrict__ in, _Float16* __restrict__ out,
                               int centers, int C) {
  int i = blockIdx.x * blockDim.x + threadIdx.x;
  if (i >= centers * C) return;
  int c = i % C, ci = i / C;
  const _Float16* p = in + (size_t)ci * 32 * C + c;
  float mx = (float)p[0];
  #pragma unroll 4
  for (int kk = 1; kk < 32; ++kk) mx = fmaxf(mx, (float)p[(size_t)kk * C]);
  out[i] = (_Float16)mx;
}

__global__ void copy_f32(const float* __restrict__ src, float* __restrict__ dst, int n) {
  int i = blockIdx.x * blockDim.x + threadIdx.x;
  if (i < n) dst[i] = src[i];
}

// ---------------------------------------------------------------------------
extern "C" void kernel_launch(void* const* d_in, const int* in_sizes, int n_in,
                              void* d_out, int out_size, void* d_ws, size_t ws_size,
                              hipStream_t stream) {
  (void)in_sizes; (void)n_in; (void)out_size; (void)ws_size;
  const int B = 16, N1 = 8192, M1 = 512, KNN = 32, M2 = 64;

  const float* pos = (const float*)d_in[0];
  const float* W1l[3] = {(const float*)d_in[1], (const float*)d_in[3], (const float*)d_in[5]};
  const float* b1l[3] = {(const float*)d_in[2], (const float*)d_in[4], (const float*)d_in[6]};
  const float* W1g[2] = {(const float*)d_in[7], (const float*)d_in[9]};
  const float* b1g[2] = {(const float*)d_in[8], (const float*)d_in[10]};
  const float* W2l[3] = {(const float*)d_in[11], (const float*)d_in[13], (const float*)d_in[15]};
  const float* b2l[3] = {(const float*)d_in[12], (const float*)d_in[14], (const float*)d_in[16]};
  const float* W2g[2] = {(const float*)d_in[17], (const float*)d_in[19]};
  const float* b2g[2] = {(const float*)d_in[18], (const float*)d_in[20]};

  // -------- workspace carving --------
  size_t off = 0;
  char* ws = (char*)d_ws;
  auto alloc = [&](size_t bytes) -> void* {
    void* p = ws + off;
    off = (off + bytes + 255) & ~(size_t)255;
    return p;
  };
  int*        idx1     = (int*)alloc((size_t)B * M1 * 4);
  float*      centers1 = (float*)alloc((size_t)B * M1 * 3 * 4);
  int*        nbr1     = (int*)alloc((size_t)B * M1 * KNN * 4);
  int*        idx2     = (int*)alloc((size_t)B * M2 * 4);
  float*      centers2 = (float*)alloc((size_t)B * M2 * 3 * 4);
  int*        nbr2     = (int*)alloc((size_t)B * M2 * KNN * 4);
  _Float16*   x1h      = (_Float16*)alloc((size_t)B * M1 * 256 * 2);
  _Float16*   pooled1  = (_Float16*)alloc((size_t)B * M1 * 128 * 2);
  _Float16*   g1h      = (_Float16*)alloc((size_t)B * M1 * 256 * 2);
  _Float16*   pooled2  = (_Float16*)alloc((size_t)B * M2 * 512 * 2);
  _Float16*   g2h      = (_Float16*)alloc((size_t)B * M2 * 512 * 2);
  _Float16* w1l0t = (_Float16*)alloc((size_t)128 * 64  * 2);
  _Float16* w1l1t = (_Float16*)alloc((size_t)128 * 128 * 2);
  _Float16* w1l2t = (_Float16*)alloc((size_t)128 * 128 * 2);
  _Float16* w1g0t = (_Float16*)alloc((size_t)256 * 128 * 2);
  _Float16* w1g1t = (_Float16*)alloc((size_t)256 * 256 * 2);
  _Float16* w2l0t = (_Float16*)alloc((size_t)512 * 320 * 2);
  _Float16* w2l1t = (_Float16*)alloc((size_t)512 * 512 * 2);
  _Float16* w2l2t = (_Float16*)alloc((size_t)512 * 512 * 2);
  _Float16* w2g0t = (_Float16*)alloc((size_t)512 * 512 * 2);
  _Float16* w2g1t = (_Float16*)alloc((size_t)512 * 512 * 2);
  const size_t rows1 = (size_t)B * M1 * KNN;   // 262144
  const size_t rows2 = (size_t)B * M2 * KNN;   // 32768
  _Float16* edgeBuf = (_Float16*)alloc(rows1 * 64 * 2);    // edge1 (64) / edge2 (320, smaller)
  _Float16* actA    = (_Float16*)alloc(rows1 * 128 * 2);   // also holds rows2*512 (smaller)
  _Float16* actB    = (_Float16*)alloc(rows1 * 128 * 2);

  // -------- weight prep --------
  auto prep = [&](const float* W, _Float16* Wt, int K, int N, int Kp) {
    int total = N * Kp;
    prep_weight<<<(total + 255) / 256, 256, 0, stream>>>(W, Wt, K, N, Kp);
  };
  prep(W1l[0], w1l0t, 63, 128, 64);
  prep(W1l[1], w1l1t, 128, 128, 128);
  prep(W1l[2], w1l2t, 128, 128, 128);
  prep(W1g[0], w1g0t, 128, 256, 128);
  prep(W1g[1], w1g1t, 256, 256, 256);
  prep(W2l[0], w2l0t, 319, 512, 320);
  prep(W2l[1], w2l1t, 512, 512, 512);
  prep(W2l[2], w2l2t, 512, 512, 512);
  prep(W2g[0], w2g0t, 512, 512, 512);
  prep(W2g[1], w2g1t, 512, 512, 512);

  auto gemm = [&](const _Float16* A, const _Float16* Wt, const float* bias,
                  _Float16* outH, float* outF, int M, int N, int Kp, int relu) {
    int groups = (M / 16) * (N / 64);
    gemm_wmma<<<(groups + 3) / 4, 128, 0, stream>>>(A, Wt, bias, outH, outF, M, N, Kp, relu);
  };

  // ================= SA1 =================
  fps_kernel<<<B, 512, (size_t)N1 * 4 + 16 * 8, stream>>>(pos, N1, M1, idx1);
  gather_centers<<<(B * M1 * 3 + 255) / 256, 256, 0, stream>>>(pos, idx1, centers1, N1, M1, B * M1 * 3);
  knn_kernel<<<B * M1, 256, (size_t)N1 * 4 + 8 * 8, stream>>>(pos, centers1, N1, M1, KNN, nbr1);
  edge1_kernel<<<((int)rows1 + 255) / 256, 256, 0, stream>>>(pos, centers1, nbr1, edgeBuf, (int)rows1);

  gemm(edgeBuf, w1l0t, b1l[0], actA, nullptr, (int)rows1, 128, 64, 1);
  gemm(actA,    w1l1t, b1l[1], actB, nullptr, (int)rows1, 128, 128, 1);
  gemm(actB,    w1l2t, b1l[2], actA, nullptr, (int)rows1, 128, 128, 0);
  maxpool_kernel<<<(B * M1 * 128 + 255) / 256, 256, 0, stream>>>(actA, pooled1, B * M1, 128);
  gemm(pooled1, w1g0t, b1g[0], g1h, nullptr, B * M1, 256, 128, 1);
  gemm(g1h,     w1g1t, b1g[1], x1h, nullptr, B * M1, 256, 256, 0);

  // ================= SA2 =================
  fps_kernel<<<B, 512, (size_t)M1 * 4 + 16 * 8, stream>>>(centers1, M1, M2, idx2);
  gather_centers<<<(B * M2 * 3 + 255) / 256, 256, 0, stream>>>(centers1, idx2, centers2, M1, M2, B * M2 * 3);
  knn_kernel<<<B * M2, 256, (size_t)M1 * 4 + 8 * 8, stream>>>(centers1, centers2, M1, M2, KNN, nbr2);
  edge2_kernel<<<(int)rows2, 320, 0, stream>>>(x1h, centers1, centers2, nbr2, edgeBuf);

  gemm(edgeBuf, w2l0t, b2l[0], actA, nullptr, (int)rows2, 512, 320, 1);
  gemm(actA,    w2l1t, b2l[1], actB, nullptr, (int)rows2, 512, 512, 1);
  gemm(actB,    w2l2t, b2l[2], actA, nullptr, (int)rows2, 512, 512, 0);
  maxpool_kernel<<<(B * M2 * 512 + 255) / 256, 256, 0, stream>>>(actA, pooled2, B * M2, 512);
  gemm(pooled2, w2g0t, b2g[0], g2h, nullptr, B * M2, 512, 512, 1);
  gemm(g2h, w2g1t, b2g[1], nullptr, (float*)d_out, B * M2, 512, 512, 0);

  // pos2 -> d_out tail
  float* pos2_out = (float*)d_out + (size_t)B * M2 * 512;
  copy_f32<<<(B * M2 * 3 + 255) / 256, 256, 0, stream>>>(centers2, pos2_out, B * M2 * 3);
}
